// TransFuser_67963562492443
// MI455X (gfx1250) — compile-verified, matrix operand order checked by reading, not run on hardware
//
#include <hip/hip_runtime.h>
#include <hip/hip_bf16.h>
#include <stdint.h>
#include <stddef.h>

// ---------------------------------------------------------------------------
// TransFuser forward for MI455X (gfx1250, wave32, WMMA).
// All GEMM-shaped math runs on v_wmma_f32_16x16x32_bf16 (bf16 in, f32 acc).
// GEMM contract (all enforced by the producers in this file):
//   - M % 16 == 0, N % 16 == 0, K % 32 == 0 (K zero-padded at production)
//   - A is [M,K] bf16 row-major, lda % 32 == 0 (or 8 for offsets), base 16B-aligned
//   - B is [N,K] bf16 K-major ("Bt"): element (k,n) at B[n*ldb + boff + k]
//   -> every fragment is two aligned global_load_b128 per lane, zero branches.
// ---------------------------------------------------------------------------

#define BN_EPS 1e-5f

typedef __attribute__((ext_vector_type(16))) unsigned short v16u;
typedef __attribute__((ext_vector_type(8)))  unsigned short v8u;
typedef __attribute__((ext_vector_type(16))) __bf16         v16bf;
typedef __attribute__((ext_vector_type(8)))  float          v8f;

__device__ __forceinline__ unsigned short f2bf(float f) {
  unsigned u = __float_as_uint(f);
  u += 0x7fffu + ((u >> 16) & 1u);   // round-to-nearest-even
  return (unsigned short)(u >> 16);
}
__device__ __forceinline__ int iclampd(int v, int lo, int hi) {
  return v < lo ? lo : (v > hi ? hi : v);
}

#define CAT16(a, b) __builtin_shufflevector(a, b, 0,1,2,3,4,5,6,7,8,9,10,11,12,13,14,15)

// ---------------------------------------------------------------------------
// WMMA GEMM. Block = 256 threads = 8 wave32 waves, 4(M) x 2(N); each wave
// computes a 16x32 tile (two 16x16 accumulators sharing the A fragment).
// Fragment layouts follow CDNA5 ISA 7.12.2 (wave32):
//   A/B 16-bit frag: lane L holds row/col (L&15); halves map to two contiguous
//   8-half runs starting at k = 8*(L>>4) and k = 8*(L>>4)+16.
//   C/D: m = r + 8*(L>>4), n = L&15.
// ---------------------------------------------------------------------------
__global__ __launch_bounds__(256) void k_gemm(
    const unsigned short* __restrict__ A, int lda, int aoff,
    const unsigned short* __restrict__ B, int ldb, int boff,
    float* __restrict__ C, int ldc, int coff,
    const float* __restrict__ bias, int relu,
    int M, int N, int K)
{
  const int lane = threadIdx.x & 31;
  const int wave = threadIdx.x >> 5;          // wave32
  const int wm = wave >> 1, wn = wave & 1;
  const int m0 = blockIdx.y * 64 + wm * 16;
  const int n0 = blockIdx.x * 64 + wn * 32;
  if (m0 >= M || n0 >= N) return;             // wave-uniform: EXEC stays full
  const bool dual = (n0 + 16) < N;            // wave-uniform
  const int l15 = lane & 15;
  const int kh  = lane >> 4;                  // chunk offset 0/1

  const v8u* __restrict__ pa  = (const v8u*)(A + (size_t)(m0 + l15) * lda + aoff);
  const v8u* __restrict__ pb0 = (const v8u*)(B + (size_t)(n0 + l15) * ldb + boff);
  const v8u* __restrict__ pb1 = (const v8u*)(B + (size_t)(n0 + 16 + l15) * ldb + boff);

  v8f acc0 = {0.f, 0.f, 0.f, 0.f, 0.f, 0.f, 0.f, 0.f};
  v8f acc1 = {0.f, 0.f, 0.f, 0.f, 0.f, 0.f, 0.f, 0.f};
  const int nch = K >> 3;                     // 16-byte chunks per row

  if (dual) {
    for (int ci = kh; ci < nch; ci += 4) {    // one k-step = 32 halves = 4 chunks
      __builtin_prefetch(pa + ci + 8, 0, 0);
      v8u a0 = pa[ci],  a1 = pa[ci + 2];
      v8u b0 = pb0[ci], b1 = pb0[ci + 2];
      v8u c0 = pb1[ci], c1 = pb1[ci + 2];
      v16u au = CAT16(a0, a1);
      v16u bu = CAT16(b0, b1);
      v16u cu = CAT16(c0, c1);
      acc0 = __builtin_amdgcn_wmma_f32_16x16x32_bf16(
          false, __builtin_bit_cast(v16bf, au), false, __builtin_bit_cast(v16bf, bu),
          (short)0, acc0, false, false);
      acc1 = __builtin_amdgcn_wmma_f32_16x16x32_bf16(
          false, __builtin_bit_cast(v16bf, au), false, __builtin_bit_cast(v16bf, cu),
          (short)0, acc1, false, false);
    }
  } else {
    for (int ci = kh; ci < nch; ci += 4) {
      v8u a0 = pa[ci],  a1 = pa[ci + 2];
      v8u b0 = pb0[ci], b1 = pb0[ci + 2];
      v16u au = CAT16(a0, a1);
      v16u bu = CAT16(b0, b1);
      acc0 = __builtin_amdgcn_wmma_f32_16x16x32_bf16(
          false, __builtin_bit_cast(v16bf, au), false, __builtin_bit_cast(v16bf, bu),
          (short)0, acc0, false, false);
    }
  }

  const int nC = n0 + l15;
  const int mr = m0 + (kh << 3);
  {
    const float bv = bias ? bias[nC] : 0.f;
#pragma unroll
    for (int r = 0; r < 8; ++r) {
      float v = acc0[r] + bv;
      if (relu) v = fmaxf(v, 0.f);
      C[(size_t)(mr + r) * ldc + coff + nC] = v;
    }
  }
  if (dual) {
    const float bv = bias ? bias[nC + 16] : 0.f;
#pragma unroll
    for (int r = 0; r < 8; ++r) {
      float v = acc1[r] + bv;
      if (relu) v = fmaxf(v, 0.f);
      C[(size_t)(mr + r) * ldc + coff + nC + 16] = v;
    }
  }
}

// --------------------------- data movement kernels -------------------------

__global__ void k_tobf(const float* __restrict__ src, unsigned short* __restrict__ dst, size_t n) {
  size_t i = (size_t)blockIdx.x * blockDim.x + threadIdx.x;
  if (i < n) dst[i] = f2bf(src[i]);
}

// transpose-convert: src [P,C] f32 -> dst [C,P] bf16
__global__ void k_tobf_t(const float* __restrict__ src, unsigned short* __restrict__ dst,
                         int P, int C) {
  size_t total = (size_t)P * C;
  size_t i = (size_t)blockIdx.x * blockDim.x + threadIdx.x;
  if (i >= total) return;
  int c = (int)(i % C); int p = (int)(i / C);
  dst[(size_t)c * P + p] = f2bf(src[i]);
}

__global__ void k_nchw2nhwc(const float* __restrict__ in, float* __restrict__ out,
                            int B, int C, int H, int W) {
  size_t i = (size_t)blockIdx.x * blockDim.x + threadIdx.x;
  size_t total = (size_t)B * C * H * W;
  if (i >= total) return;
  int c = (int)(i % C); size_t t = i / C;
  int x = (int)(t % W); t /= W;
  int y = (int)(t % H); int b = (int)(t / H);
  out[i] = in[(((size_t)b * C + c) * H + y) * W + x];
}

// im2col: col[p*Kp + ((ky*KW+kx)*C + ci)] from NHWC-flat activation, bf16 out,
// rows padded with zeros up to Kp (multiple of 32).
__global__ void k_im2col(const float* __restrict__ act, unsigned short* __restrict__ col,
                         int B, int H, int W, int C, int KH, int KW,
                         int stride, int pad, int OH, int OW, int Kp) {
  size_t total = (size_t)B * OH * OW * Kp;
  size_t i = (size_t)blockIdx.x * blockDim.x + threadIdx.x;
  if (i >= total) return;
  int kidx = (int)(i % Kp); size_t p = i / Kp;
  int Kdim = KH * KW * C;
  if (kidx >= Kdim) { col[i] = 0; return; }
  int ci = kidx % C; int kk = kidx / C;
  int kx = kk % KW;  int ky = kk / KW;
  int ox = (int)(p % OW); size_t t = p / OW;
  int oy = (int)(t % OH); int b = (int)(t / OH);
  int iy = oy * stride - pad + ky, ix = ox * stride - pad + kx;
  float v = 0.f;
  if (iy >= 0 && iy < H && ix >= 0 && ix < W)
    v = act[(((size_t)b * H + iy) * W + ix) * C + ci];
  col[i] = f2bf(v);
}

// OIHW -> Bt [Cout, Kp] with k = (ky*KW+kx)*Cin+ci, zero-padded to Kp
__global__ void k_wconv(const float* __restrict__ w, unsigned short* __restrict__ wq,
                        int Cout, int Cin, int KH, int KW, int Kp) {
  size_t total = (size_t)Cout * Kp;
  size_t i = (size_t)blockIdx.x * blockDim.x + threadIdx.x;
  if (i >= total) return;
  int k = (int)(i % Kp); int co = (int)(i / Kp);
  int Kdim = KH * KW * Cin;
  float v = 0.f;
  if (k < Kdim) {
    int ci = k % Cin; int kk = k / Cin;
    int kx = kk % KW; int ky = kk / KW;
    v = w[(((size_t)co * Cin + ci) * KH + ky) * KW + kx];
  }
  wq[i] = f2bf(v);
}

// linear weight -> Bt [N, Kp] ; trans: src[(rowoff+n)*K + k], else src[k*N+n]
__global__ void k_wlin(const float* __restrict__ w, unsigned short* __restrict__ wq,
                       int K, int N, int trans, int rowoff, int Kp) {
  size_t total = (size_t)N * Kp;
  size_t i = (size_t)blockIdx.x * blockDim.x + threadIdx.x;
  if (i >= total) return;
  int k = (int)(i % Kp); int n = (int)(i / Kp);
  float v = 0.f;
  if (k < K) v = trans ? w[(size_t)(rowoff + n) * K + k] : w[(size_t)k * N + n];
  wq[i] = f2bf(v);
}

// ------------------------------- norms etc ---------------------------------

__global__ void k_bn_stats(const float* __restrict__ x, float* __restrict__ mean,
                           float* __restrict__ var, int P, int C) {
  int c = blockIdx.x; int t = threadIdx.x;
  float s = 0.f, s2 = 0.f;
  for (int p = t; p < P; p += 256) {
    float v = x[(size_t)p * C + c]; s += v; s2 += v * v;
  }
  __shared__ float rs[256], rq[256];
  rs[t] = s; rq[t] = s2; __syncthreads();
  for (int o = 128; o > 0; o >>= 1) {
    if (t < o) { rs[t] += rs[t + o]; rq[t] += rq[t + o]; }
    __syncthreads();
  }
  if (t == 0) { float m = rs[0] / P; mean[c] = m; var[c] = rq[0] / P - m * m; }
}

__global__ void k_bn_apply(float* __restrict__ y, const float* __restrict__ x,
                           const float* __restrict__ mean, const float* __restrict__ var,
                           const float* __restrict__ g, const float* __restrict__ b,
                           const float* __restrict__ resid, int relu,
                           size_t total, int C) {
  size_t i = (size_t)blockIdx.x * blockDim.x + threadIdx.x;
  if (i >= total) return;
  int c = (int)(i % C);
  float v = g[c] * (x[i] - mean[c]) * rsqrtf(var[c] + BN_EPS) + b[c];
  if (resid) v += resid[i];
  if (relu) v = fmaxf(v, 0.f);
  y[i] = v;
}

// x = LN(a + b2) row-wise, C==256==blockDim
__global__ void k_add_ln(float* __restrict__ out, const float* __restrict__ a,
                         const float* __restrict__ b2,
                         const float* __restrict__ g, const float* __restrict__ beta) {
  const int C = 256;
  int row = blockIdx.x, t = threadIdx.x;
  size_t idx = (size_t)row * C + t;
  float v = a[idx] + b2[idx];
  __shared__ float red[256];
  red[t] = v; __syncthreads();
  for (int o = 128; o > 0; o >>= 1) { if (t < o) red[t] += red[t + o]; __syncthreads(); }
  float m = red[0] / C; __syncthreads();
  float d = v - m;
  red[t] = d * d; __syncthreads();
  for (int o = 128; o > 0; o >>= 1) { if (t < o) red[t] += red[t + o]; __syncthreads(); }
  float var = red[0] / C;
  out[idx] = g[t] * d * rsqrtf(var + BN_EPS) + beta[t];
}

// row softmax(s*scale) -> bf16 probs ; one block per row
__global__ void k_softmax(const float* __restrict__ s, unsigned short* __restrict__ p,
                          int cols, float scale) {
  int row = blockIdx.x, t = threadIdx.x;
  const float* r = s + (size_t)row * cols;
  __shared__ float red[256];
  float mx = -3.4e38f;
  for (int i = t; i < cols; i += 256) mx = fmaxf(mx, r[i] * scale);
  red[t] = mx; __syncthreads();
  for (int o = 128; o > 0; o >>= 1) { if (t < o) red[t] = fmaxf(red[t], red[t + o]); __syncthreads(); }
  mx = red[0]; __syncthreads();
  float sum = 0.f;
  for (int i = t; i < cols; i += 256) sum += __expf(r[i] * scale - mx);
  red[t] = sum; __syncthreads();
  for (int o = 128; o > 0; o >>= 1) { if (t < o) red[t] += red[t + o]; __syncthreads(); }
  float inv = 1.f / red[0];
  for (int i = t; i < cols; i += 256)
    p[(size_t)row * cols + i] = f2bf(__expf(r[i] * scale - mx) * inv);
}

__global__ void k_maxpool(const float* __restrict__ in, float* __restrict__ out,
                          int B, int H, int W, int C, int OH, int OW) {
  size_t total = (size_t)B * OH * OW * C;
  size_t i = (size_t)blockIdx.x * blockDim.x + threadIdx.x;
  if (i >= total) return;
  int c = (int)(i % C); size_t t = i / C;
  int ox = (int)(t % OW); t /= OW;
  int oy = (int)(t % OH); int b = (int)(t / OH);
  float m = -3.4e38f;
  for (int dy = 0; dy < 3; ++dy)
    for (int dx = 0; dx < 3; ++dx) {
      int iy = oy * 2 - 1 + dy, ix = ox * 2 - 1 + dx;
      if (iy >= 0 && iy < H && ix >= 0 && ix < W)
        m = fmaxf(m, in[(((size_t)b * H + iy) * W + ix) * C + c]);
    }
  out[i] = m;
}

// ---------------------------- lidar branch ---------------------------------

// normalized points, written with row stride 32 (zero padded) for the K=32 GEMM
__global__ void k_ptprep(const float* __restrict__ pts, unsigned short* __restrict__ bf, int Np) {
  int i = blockIdx.x * blockDim.x + threadIdx.x;
  if (i >= Np) return;
  const float* p = pts + (size_t)i * 7;
  unsigned short* d = bf + (size_t)i * 32;
  d[0] = f2bf((p[0] + 51.2f) / 1.6f);
  d[1] = f2bf((p[1] + 51.2f) / 1.6f);
  d[2] = f2bf((p[2] + 3.0f) / 0.6f);
  d[3] = f2bf(p[3]); d[4] = f2bf(p[4]); d[5] = f2bf(p[5]); d[6] = f2bf(p[6]);
#pragma unroll
  for (int j = 7; j < 32; ++j) d[j] = 0;
}

__global__ void k_fill_u32(unsigned int* __restrict__ p, unsigned int v, size_t n) {
  size_t i = (size_t)blockIdx.x * blockDim.x + threadIdx.x;
  if (i < n) p[i] = v;
}

__global__ void k_scatter(const float* __restrict__ pts, const float* __restrict__ feats,
                          unsigned int* __restrict__ bevU, int Np, int D) {
  size_t total = (size_t)Np * D;
  size_t i = (size_t)blockIdx.x * blockDim.x + threadIdx.x;
  if (i >= total) return;
  int d = (int)(i % D); int n = (int)(i / D);
  const float* p = pts + (size_t)n * 7;
  int gx = iclampd((int)((p[0] + 51.2f) / 1.6f), 0, 63);
  int gy = iclampd((int)((p[1] + 51.2f) / 1.6f), 0, 63);
  unsigned u = __float_as_uint(feats[i]);
  u = (u & 0x80000000u) ? ~u : (u | 0x80000000u);   // monotone float->uint
  atomicMax(&bevU[(size_t)(gx * 64 + gy) * D + d], u);
}

__global__ void k_scatter_decode(const unsigned int* __restrict__ bevU,
                                 unsigned short* __restrict__ bevBF, size_t n) {
  size_t i = (size_t)blockIdx.x * blockDim.x + threadIdx.x;
  if (i >= n) return;
  unsigned u = bevU[i];
  unsigned bits = (u & 0x80000000u) ? (u ^ 0x80000000u) : ~u;
  bevBF[i] = f2bf(__uint_as_float(bits));
}

// ---------------------------- misc glue ------------------------------------

__global__ void k_mean3(const float* __restrict__ in, float* __restrict__ out, int n) {
  int i = blockIdx.x * blockDim.x + threadIdx.x;
  if (i >= n) return;
  out[i] = (in[i] + in[i + n] + in[i + 2 * n]) * (1.f / 3.f);
}

__global__ void k_bilinear(const float* __restrict__ in, float* __restrict__ out,
                           int Hin, int Win, int Hout, int Wout, int C) {
  size_t total = (size_t)Hout * Wout * C;
  size_t i = (size_t)blockIdx.x * blockDim.x + threadIdx.x;
  if (i >= total) return;
  int c = (int)(i % C); size_t t = i / C;
  int x = (int)(t % Wout); int y = (int)(t / Wout);
  float sy = (y + 0.5f) * (float)Hin / (float)Hout - 0.5f;
  float sx = (x + 0.5f) * (float)Win / (float)Wout - 0.5f;
  int y0 = (int)floorf(sy), x0 = (int)floorf(sx);
  float fy = sy - y0, fx = sx - x0;
  int y0c = iclampd(y0, 0, Hin - 1), y1c = iclampd(y0 + 1, 0, Hin - 1);
  int x0c = iclampd(x0, 0, Win - 1), x1c = iclampd(x0 + 1, 0, Win - 1);
  float v00 = in[((size_t)y0c * Win + x0c) * C + c];
  float v01 = in[((size_t)y0c * Win + x1c) * C + c];
  float v10 = in[((size_t)y1c * Win + x0c) * C + c];
  float v11 = in[((size_t)y1c * Win + x1c) * C + c];
  out[i] = (1.f - fy) * ((1.f - fx) * v00 + fx * v01) + fy * ((1.f - fx) * v10 + fx * v11);
}

__global__ void k_transpose_pc_cp(const float* __restrict__ in, float* __restrict__ out,
                                  int P, int C) {
  size_t total = (size_t)P * C;
  size_t i = (size_t)blockIdx.x * blockDim.x + threadIdx.x;
  if (i >= total) return;
  int c = (int)(i % C); int p = (int)(i / C);
  out[(size_t)c * P + p] = in[i];
}

// imf [3*8*16, 256] NHWC-flat -> d_out region as (3,256,8,16)
__global__ void k_imgfeat(const float* __restrict__ imf, float* __restrict__ out) {
  int i = blockIdx.x * blockDim.x + threadIdx.x;
  if (i >= 98304) return;
  int x = i % 16; int t = i / 16;
  int y = t % 8; t /= 8;
  int c = t % 256; int n = t / 256;
  out[i] = imf[(((size_t)n * 8 + y) * 16 + x) * 256 + c];
}

// ---------------------------------------------------------------------------

extern "C" void kernel_launch(void* const* d_in, const int* in_sizes, int n_in,
                              void* d_out, int out_size, void* d_ws, size_t ws_size,
                              hipStream_t stream) {
  (void)in_sizes; (void)n_in; (void)out_size; (void)ws_size;
  const float* images = (const float*)d_in[0];
  const float* lidar  = (const float*)d_in[1];
  float* out = (float*)d_out;

  // ---- params: jax pytree flatten order (sorted dict keys) ----
  int cur = 2;
  auto nextp = [&]() { return (const float*)d_in[cur++]; };
  const float* img_b = nextp(); const float* img_w = nextp();
  struct LayerP { const float *f1b,*f1w,*f2b,*f2w,*in_b,*in_w,*ln1b,*ln1g,*ln2b,*ln2g,*out_b,*out_w; } L[2];
  for (int l = 0; l < 2; ++l) {
    L[l].f1b = nextp(); L[l].f1w = nextp(); L[l].f2b = nextp(); L[l].f2w = nextp();
    L[l].in_b = nextp(); L[l].in_w = nextp();
    L[l].ln1b = nextp(); L[l].ln1g = nextp(); L[l].ln2b = nextp(); L[l].ln2g = nextp();
    L[l].out_b = nextp(); L[l].out_w = nextp();
  }
  const float* lid_b = nextp(); const float* lid_w = nextp();
  const float *m_b1 = nextp(), *m_b2 = nextp(), *m_b3 = nextp(),
              *m_bb1 = nextp(), *m_bb2 = nextp(),
              *m_g1 = nextp(), *m_g2 = nextp(),
              *m_w1 = nextp(), *m_w2 = nextp(), *m_w3 = nextp();
  const float* fout_b = nextp(); const float* fout_w = nextp();
  const float* proj_b = nextp(); const float* proj_w = nextp();
  struct BlkP { const float *b1,*b2,*bd,*g1,*g2,*gd,*w1,*w2,*wd; bool down; } blk[16];
  { int nb = 0; const int cnt[4] = {3, 4, 6, 3};
    for (int s = 0; s < 4; ++s)
      for (int i = 0; i < cnt[s]; ++i) {
        BlkP b{}; b.down = (s > 0 && i == 0);
        b.b1 = nextp(); b.b2 = nextp(); if (b.down) b.bd = nextp();
        b.g1 = nextp(); b.g2 = nextp(); if (b.down) b.gd = nextp();
        b.w1 = nextp(); b.w2 = nextp(); if (b.down) b.wd = nextp();
        blk[nb++] = b;
      } }
  const float* stem_b = nextp(); const float* stem_g = nextp(); const float* stem_w = nextp();

  // ---- workspace bump allocator (needs ~320 MB) ----
  char* wsb = (char*)d_ws; size_t off = 0;
  auto alloc = [&](size_t bytes) -> void* {
    void* p = wsb + off; off = (off + bytes + 255) & ~(size_t)255; return p;
  };
  float* bufIn  = (float*)alloc((size_t)1179648 * 4);           // [3,256,512,3]
  float* stemF  = (float*)alloc((size_t)6291456 * 4);           // [3,128,256,64]
  float* arena0 = (float*)alloc((size_t)1600000 * 4);
  float* arena1 = (float*)alloc((size_t)1600000 * 4);
  float* arena2 = (float*)alloc((size_t)1600000 * 4);
  float* arena3 = (float*)alloc((size_t)1600000 * 4);
  unsigned short* colBF = (unsigned short*)alloc((size_t)16000000 * 2); // padded im2col
  unsigned short* wqBF  = (unsigned short*)alloc((size_t)2400000 * 2);
  float* meanF = (float*)alloc(1024 * 4);
  float* varF  = (float*)alloc(1024 * 4);
  unsigned short* bigBF = (unsigned short*)alloc((size_t)15500000 * 2); // serial bf16 scratch
  char* bigRaw = (char*)alloc((size_t)135 * 1024 * 1024);       // serial f32 scratch
  float* h1F     = (float*)bigRaw;                               // [120000,64]
  float* h2F     = (float*)(bigRaw + (size_t)35 * 1024 * 1024);  // [120000,128]
  float* featsF  = (float*)bigRaw;                               // [120000,256]
  float* scoresF = (float*)bigRaw;                               // [4096,4096]
  unsigned short* probsBF = (unsigned short*)(bigRaw + (size_t)70 * 1024 * 1024);
  unsigned short* qBF  = (unsigned short*)alloc((size_t)1048576 * 2);
  unsigned short* kvBF = (unsigned short*)alloc((size_t)1048576 * 2);
  unsigned short* qpBF = (unsigned short*)alloc((size_t)1048576 * 2);
  unsigned short* kpBF = (unsigned short*)alloc((size_t)1048576 * 2);
  unsigned short* vpBF = (unsigned short*)alloc((size_t)1048576 * 2);  // stored as [256,4096]
  unsigned int* bevU = (unsigned int*)alloc((size_t)1048576 * 4);
  float* lidarBevF = (float*)alloc((size_t)1048576 * 4);
  float* fusedF = (float*)alloc((size_t)1048576 * 4);
  float* xF     = (float*)alloc((size_t)1048576 * 4);
  float* tmpF   = (float*)alloc((size_t)1048576 * 4);
  float* attnF  = (float*)alloc((size_t)1048576 * 4);
  float* ffF    = (float*)alloc((size_t)4194304 * 4);
  float* imfF   = (float*)alloc((size_t)98304 * 4);
  float* imgPI  = (float*)alloc((size_t)98304 * 4);
  float* imgSm  = (float*)alloc((size_t)32768 * 4);

  // ---- host helpers ----
  auto cdiv = [](size_t a, size_t b) -> unsigned { return (unsigned)((a + b - 1) / b); };
  auto tobf = [&](unsigned short* dst, const float* src, size_t n) {
    k_tobf<<<cdiv(n, 256), 256, 0, stream>>>(src, dst, n);
  };
  auto gemm = [&](const unsigned short* A, int lda, int aoff,
                  const unsigned short* B, int ldb, int boff,
                  float* C, int ldc, int coff, const float* bias, int relu,
                  int M, int N, int K) {
    dim3 g((unsigned)((N + 63) / 64), (unsigned)((M + 63) / 64));
    k_gemm<<<g, 256, 0, stream>>>(A, lda, aoff, B, ldb, boff,
                                  C, ldc, coff, bias, relu, M, N, K);
  };
  auto wlin = [&](const float* w, int K, int N, int trans, int rowoff) -> int {
    int Kp = (K + 31) & ~31;
    k_wlin<<<cdiv((size_t)N * Kp, 256), 256, 0, stream>>>(w, wqBF, K, N, trans, rowoff, Kp);
    return Kp;
  };
  auto bn = [&](float* y, const float* x, const float* g, const float* b,
                const float* resid, int relu, size_t P, int C) {
    k_bn_stats<<<C, 256, 0, stream>>>(x, meanF, varF, (int)P, C);
    k_bn_apply<<<cdiv(P * (size_t)C, 256), 256, 0, stream>>>(
        y, x, meanF, varF, g, b, resid, relu, P * (size_t)C, C);
  };
  auto conv = [&](const float* actF, int B, int H, int W, int Cin,
                  const float* w, int Cout, int KH, int KW, int stride, int pad,
                  float* outF, int OH, int OW) {
    int Kp = (KH * KW * Cin + 31) & ~31;
    size_t Pout = (size_t)B * OH * OW;
    k_im2col<<<cdiv(Pout * Kp, 256), 256, 0, stream>>>(
        actF, colBF, B, H, W, Cin, KH, KW, stride, pad, OH, OW, Kp);
    k_wconv<<<cdiv((size_t)Cout * Kp, 256), 256, 0, stream>>>(w, wqBF, Cout, Cin, KH, KW, Kp);
    gemm(colBF, Kp, 0, wqBF, Kp, 0, outF, Cout, 0, nullptr, 0, (int)Pout, Cout, Kp);
  };

  // =================== image branch: ResNet34 (train-mode BN) ===============
  k_nchw2nhwc<<<cdiv((size_t)3 * 3 * 256 * 512, 256), 256, 0, stream>>>(images, bufIn, 3, 3, 256, 512);
  conv(bufIn, 3, 256, 512, 3, stem_w, 64, 7, 7, 2, 3, stemF, 128, 256);
  bn(stemF, stemF, stem_g, stem_b, nullptr, 1, (size_t)3 * 128 * 256, 64);
  k_maxpool<<<cdiv((size_t)3 * 64 * 128 * 64, 256), 256, 0, stream>>>(stemF, arena0, 3, 128, 256, 64, 64, 128);

  float* X = arena0; int H = 64, W = 128, Cin = 64;
  { int bi = 0; const int cnt[4] = {3, 4, 6, 3};
    const int couts[4] = {64, 128, 256, 512}; const int strd[4] = {1, 2, 2, 2};
    for (int s = 0; s < 4; ++s)
      for (int i = 0; i < cnt[s]; ++i) {
        BlkP& b = blk[bi++];
        int st = (i == 0) ? strd[s] : 1; int Cout = couts[s];
        int OH = H / st, OW = W / st;
        float* Y1 = arena2;
        conv(X, 3, H, W, Cin, b.w1, Cout, 3, 3, st, 1, Y1, OH, OW);
        bn(Y1, Y1, b.g1, b.b1, nullptr, 1, (size_t)3 * OH * OW, Cout);
        float* Y2 = (X == arena0) ? arena1 : arena0;
        conv(Y1, 3, OH, OW, Cout, b.w2, Cout, 3, 3, 1, 1, Y2, OH, OW);
        const float* idn = X;
        if (b.down) {
          conv(X, 3, H, W, Cin, b.wd, Cout, 1, 1, st, 0, arena3, OH, OW);
          bn(arena3, arena3, b.gd, b.bd, nullptr, 0, (size_t)3 * OH * OW, Cout);
          idn = arena3;
        }
        bn(Y2, Y2, b.g2, b.b2, idn, 1, (size_t)3 * OH * OW, Cout); // relu(bn+idn)
        X = Y2; H = OH; W = OW; Cin = Cout;
      } }

  // proj 1x1 -> imf [384,256]
  const int Pimg = 384;
  tobf(bigBF, X, (size_t)Pimg * 512);
  wlin(proj_w, 512, 256, 1, 0);
  gemm(bigBF, 512, 0, wqBF, 512, 0, imfF, 256, 0, proj_b, 0, Pimg, 256, 512);
  k_imgfeat<<<cdiv(98304, 256), 256, 0, stream>>>(imfF, out + 1048576);

  // image BEV: img 1x1 conv, mean over cameras, bilinear 8x16 -> 64x64
  tobf(bigBF, imfF, (size_t)Pimg * 256);
  wlin(img_w, 256, 256, 1, 0);
  gemm(bigBF, 256, 0, wqBF, 256, 0, imgPI, 256, 0, img_b, 0, Pimg, 256, 256);
  k_mean3<<<cdiv(32768, 256), 256, 0, stream>>>(imgPI, imgSm, 32768);
  k_bilinear<<<cdiv((size_t)4096 * 256, 256), 256, 0, stream>>>(imgSm, fusedF, 8, 16, 64, 64, 256);

  // =================== lidar branch: point MLP + scatter-max =================
  k_ptprep<<<cdiv(120000, 256), 256, 0, stream>>>(lidar, bigBF, 120000);
  wlin(m_w1, 7, 64, 0, 0);                                  // Kp = 32
  gemm(bigBF, 32, 0, wqBF, 32, 0, h1F, 64, 0, m_b1, 0, 120000, 64, 32);
  bn(h1F, h1F, m_g1, m_bb1, nullptr, 1, 120000, 64);
  tobf(bigBF, h1F, (size_t)120000 * 64);
  wlin(m_w2, 64, 128, 0, 0);
  gemm(bigBF, 64, 0, wqBF, 64, 0, h2F, 128, 0, m_b2, 0, 120000, 128, 64);
  bn(h2F, h2F, m_g2, m_bb2, nullptr, 1, 120000, 128);
  tobf(bigBF, h2F, (size_t)120000 * 128);
  wlin(m_w3, 128, 256, 0, 0);
  gemm(bigBF, 128, 0, wqBF, 128, 0, featsF, 256, 0, m_b3, 0, 120000, 256, 128);
  k_fill_u32<<<cdiv((size_t)1048576, 256), 256, 0, stream>>>(bevU, 0x80000000u, 1048576);
  k_scatter<<<cdiv((size_t)120000 * 256, 256), 256, 0, stream>>>(lidar, featsF, bevU, 120000, 256);
  k_scatter_decode<<<cdiv((size_t)1048576, 256), 256, 0, stream>>>(bevU, bigBF, 1048576);
  wlin(lid_w, 256, 256, 1, 0);
  gemm(bigBF, 256, 0, wqBF, 256, 0, lidarBevF, 256, 0, lid_b, 0, 4096, 256, 256);
  k_transpose_pc_cp<<<cdiv((size_t)1048576, 256), 256, 0, stream>>>(lidarBevF, out + 1146880, 4096, 256);

  // =================== cross-attention fusion (2 layers) =====================
  tobf(kvBF, lidarBevF, (size_t)1048576);
  const float scale = 0.1767766953f;   // 1/sqrt(32)
  for (int l = 0; l < 2; ++l) {
    LayerP& Lp = L[l];
    tobf(qBF, fusedF, (size_t)1048576);
    wlin(Lp.in_w, 256, 256, 1, 0);
    gemm(qBF, 256, 0, wqBF, 256, 0, tmpF, 256, 0, Lp.in_b, 0, 4096, 256, 256);
    tobf(qpBF, tmpF, (size_t)1048576);
    wlin(Lp.in_w, 256, 256, 1, 256);
    gemm(kvBF, 256, 0, wqBF, 256, 0, tmpF, 256, 0, Lp.in_b + 256, 0, 4096, 256, 256);
    tobf(kpBF, tmpF, (size_t)1048576);
    wlin(Lp.in_w, 256, 256, 1, 512);
    gemm(kvBF, 256, 0, wqBF, 256, 0, tmpF, 256, 0, Lp.in_b + 512, 0, 4096, 256, 256);
    k_tobf_t<<<cdiv((size_t)1048576, 256), 256, 0, stream>>>(tmpF, vpBF, 4096, 256); // V^T [256,4096]
    for (int h = 0; h < 8; ++h) {
      // scores = Q_h (A, k-contig) x K_h^T (Bt, k-contig)
      gemm(qpBF, 256, h * 32, kpBF, 256, h * 32, scoresF, 4096, 0, nullptr, 0, 4096, 4096, 32);
      k_softmax<<<4096, 256, 0, stream>>>(scoresF, probsBF, 4096, scale);
      // attn_h = P (A, lda=4096) x V_h (Bt rows n -> vpT[(h*32+n)*4096 + k])
      gemm(probsBF, 4096, 0, vpBF + (size_t)h * 32 * 4096, 4096, 0,
           attnF, 256, h * 32, nullptr, 0, 4096, 32, 4096);
    }
    tobf(bigBF, attnF, (size_t)1048576);
    wlin(Lp.out_w, 256, 256, 1, 0);
    gemm(bigBF, 256, 0, wqBF, 256, 0, tmpF, 256, 0, Lp.out_b, 0, 4096, 256, 256);
    k_add_ln<<<4096, 256, 0, stream>>>(xF, fusedF, tmpF, Lp.ln1g, Lp.ln1b);
    tobf(bigBF, xF, (size_t)1048576);
    wlin(Lp.f1w, 256, 1024, 0, 0);
    gemm(bigBF, 256, 0, wqBF, 256, 0, ffF, 1024, 0, Lp.f1b, 1, 4096, 1024, 256);
    tobf(bigBF, ffF, (size_t)4194304);
    wlin(Lp.f2w, 1024, 256, 0, 0);
    gemm(bigBF, 1024, 0, wqBF, 1024, 0, tmpF, 256, 0, Lp.f2b, 0, 4096, 256, 1024);
    k_add_ln<<<4096, 256, 0, stream>>>(fusedF, xF, tmpF, Lp.ln2g, Lp.ln2b);
  }

  // =================== output head ===========================================
  tobf(bigBF, fusedF, (size_t)1048576);
  wlin(fout_w, 256, 256, 1, 0);
  gemm(bigBF, 256, 0, wqBF, 256, 0, tmpF, 256, 0, fout_b, 0, 4096, 256, 256);
  k_transpose_pc_cp<<<cdiv((size_t)1048576, 256), 256, 0, stream>>>(tmpF, out, 4096, 256);
}
